// PPNP_PYG_71854802862592
// MI455X (gfx1250) — matrix operand order repaired and checked by reading, not compiled, not run on hardware
//
#include <hip/hip_runtime.h>
#include <hip/hip_bf16.h>

#define N_NODES 100000
#define N_EDGES 3200000
#define F_INF   1024
#define HIDF    256
#define NCLS    19
#define KPROP   10
#define ALPHA_F 0.1f

typedef __bf16 bf16;
typedef bf16  v16bf __attribute__((ext_vector_type(16)));
typedef bf16  v8bf  __attribute__((ext_vector_type(8)));
typedef bf16  v4bf  __attribute__((ext_vector_type(4)));
typedef float v8f   __attribute__((ext_vector_type(8)));
typedef int   v4i   __attribute__((ext_vector_type(4)));

#if defined(__has_builtin)
#if __has_builtin(__builtin_amdgcn_global_load_async_to_lds_b128)
#define HAVE_ASYNC_LDS 1
#endif
#endif

__device__ __forceinline__ v16bf cat8(v8bf lo, v8bf hi) {
  return __builtin_shufflevector(lo, hi, 0,1,2,3,4,5,6,7,8,9,10,11,12,13,14,15);
}

// ---------------------------------------------------------------------------
// One-time: W1 [1024,256] f32 -> w1t bf16, per-K-panel contiguous:
//   w1t[p*8192 + n*32 + kk] = W1[(p*32+kk)*256 + n]
// so each K-step's B panel (256 cols x 32 k) is one contiguous 16KB block.
// ---------------------------------------------------------------------------
__global__ void cvt_w1_kernel(const float* __restrict__ W1, bf16* __restrict__ w1t) {
  int idx = blockIdx.x * blockDim.x + threadIdx.x;
  if (idx >= F_INF * HIDF) return;
  int p  = idx >> 13;          // panel
  int r  = idx & 8191;
  int n  = r >> 5;
  int kk = r & 31;
  w1t[idx] = (bf16)W1[(size_t)(p * 32 + kk) * HIDF + n];
}

// ---------------------------------------------------------------------------
// GEMM1: h1 = relu(x @ W1 + b1)   [N,1024] x [1024,256] -> bf16 [N,256]
// block = 256 thr (8 waves), tile 64x256, K-step 32, bf16 WMMA w/ f32 accum
// B panel arrives via global_load_async_to_lds_b128 (ASYNCcnt); A staged
// through VGPRs (needs f32->bf16 cvt) and overlaps the async copy.
// ---------------------------------------------------------------------------
__global__ __launch_bounds__(256) void gemm1_relu_kernel(
    const float* __restrict__ x, const bf16* __restrict__ w1t,
    const float* __restrict__ b1, bf16* __restrict__ h1) {
  __shared__ bf16 As[64][40];    // A tile 64x32 bf16, rows padded to 80B
  __shared__ bf16 Bs[256][40];   // B tile transposed: Bs[n][k] (k<32)

  const int tid = threadIdx.x;
  const int blockRow = blockIdx.x * 64;
  const int w = tid >> 5, lane = tid & 31;
  const int mt = w >> 1;             // M-subtile 0..3 (16 rows each)
  const int ngrp = (w & 1) * 8;      // first of 8 N-tiles for this wave
  const int mlane = lane & 15, hi = lane >> 4;

  v8f acc[8] = {};

  for (int k0 = 0; k0 < F_INF; k0 += 32) {
    // --- stage B: contiguous 16KB bf16 panel -> LDS, 4 unconditional issues
    const bf16* panel = w1t + (size_t)(k0 >> 5) * (HIDF * 32);
    #pragma unroll
    for (int ii = 0; ii < 4; ++ii) {
      int i = ii * 256 + tid;                        // 0..1023, 16B chunks
      int n = i >> 2, kk8 = (i & 3) << 3;
#if HAVE_ASYNC_LDS
      __builtin_amdgcn_global_load_async_to_lds_b128(
          (v4i*)(panel + (size_t)i * 8), (v4i*)&Bs[n][kk8], 0, 0);
#else
      *(v8bf*)&Bs[n][kk8] = *(const v8bf*)(panel + (size_t)i * 8);
#endif
    }

    // --- stage A: 64x32 f32 -> bf16. Phase 1: issue BOTH float4 loads
    // (distinct dest regs -> one loadcnt wait), Phase 2: convert + store.
    float4 va[2];
    int   ar[2], ac4[2];
    #pragma unroll
    for (int ii = 0; ii < 2; ++ii) {
      int i = ii * 256 + tid;                        // 0..511 float4 chunks
      ar[ii]  = i >> 3;
      ac4[ii] = (i & 7) << 2;
      int gr = blockRow + ar[ii];
      if (gr >= N_NODES) gr = N_NODES - 1;           // clamp; stores guarded
      va[ii] = *(const float4*)(x + (size_t)gr * F_INF + k0 + ac4[ii]);
    }
    #pragma unroll
    for (int ii = 0; ii < 2; ++ii) {
      v4bf o = { (bf16)va[ii].x, (bf16)va[ii].y, (bf16)va[ii].z, (bf16)va[ii].w };
      *(v4bf*)&As[ar[ii]][ac4[ii]] = o;
    }
    if (k0 + 32 < F_INF)
      __builtin_prefetch(x + (size_t)(blockRow + (tid & 63)) * F_INF + k0 + 32, 0, 0);

#if HAVE_ASYNC_LDS
    asm volatile("s_wait_asynccnt 0" ::: "memory");
#endif
    __syncthreads();

    // A fragment: lane (m,hi) -> K = {8*hi..8*hi+7, 16+8*hi..16+8*hi+7}
    const bf16* arow = &As[mt * 16 + mlane][0];
    v16bf a = cat8(*(const v8bf*)(arow + 8 * hi),
                   *(const v8bf*)(arow + 16 + 8 * hi));

    // B fragment: lane (n,hi) -> K = {8*hi..8*hi+7, 16+8*hi..16+8*hi+7} of column n.
    auto loadB = [&](int j) -> v16bf {
      const bf16* brow = &Bs[(ngrp + j) * 16 + mlane][8 * hi];
      return cat8(*(const v8bf*)brow, *(const v8bf*)(brow + 16));
    };
    v16bf b = loadB(0);
    #pragma unroll
    for (int j = 0; j < 8; ++j) {
      v16bf bn;
      if (j < 7) bn = loadB(j + 1);
      acc[j] = __builtin_amdgcn_wmma_f32_16x16x32_bf16(
          false, a, false, b, (short)0, acc[j], false, false);
      if (j < 7) b = bn;
    }
    __syncthreads();
  }

  // bias + relu + store bf16 (C layout: VGPR r -> row r+8*hi, lane&15 -> col)
  #pragma unroll
  for (int j = 0; j < 8; ++j) {
    int ncol = (ngrp + j) * 16 + mlane;
    float bias = b1[ncol];
    #pragma unroll
    for (int r = 0; r < 8; ++r) {
      int gr = blockRow + mt * 16 + r + 8 * hi;
      if (gr < N_NODES) {
        float v = acc[j][r] + bias;
        h1[(size_t)gr * HIDF + ncol] = (bf16)fmaxf(v, 0.0f);
      }
    }
  }
}

// ---------------------------------------------------------------------------
// GEMM2: h0 = h1 @ W2 + b2   [N,256](bf16) x [256,19] -> f32 [N,19] (pad N->32)
// ---------------------------------------------------------------------------
__global__ __launch_bounds__(256) void gemm2_kernel(
    const bf16* __restrict__ h1, const float* __restrict__ W2,
    const float* __restrict__ b2, float* __restrict__ h0) {
  __shared__ bf16 Bs[32][264];   // Bs[n][k]; n padded 19->32 (clamped dup cols)

  const int tid = threadIdx.x;
  #pragma unroll
  for (int ii = 0; ii < 32; ++ii) {
    int i = ii * 256 + tid;
    int n = i >> 8, k = i & 255;
    int nc = (n < NCLS) ? n : (NCLS - 1);       // clamp: padded cols never stored
    Bs[n][k] = (bf16)W2[(size_t)k * NCLS + nc];
  }
  __syncthreads();

  const int w = tid >> 5, lane = tid & 31;
  const int mlane = lane & 15, hi = lane >> 4;
  const int rowBase = blockIdx.x * 128 + w * 16;
  int gm = rowBase + mlane;
  if (gm >= N_NODES) gm = N_NODES - 1;          // clamp (stores are guarded)
  const bf16* arow = h1 + (size_t)gm * HIDF;

  v8f acc0 = {0,0,0,0,0,0,0,0};
  v8f acc1 = {0,0,0,0,0,0,0,0};
  #pragma unroll
  for (int s = 0; s < 8; ++s) {
    int k0 = s * 32;
    v16bf a = cat8(*(const v8bf*)(arow + k0 + 8 * hi),
                   *(const v8bf*)(arow + k0 + 16 + 8 * hi));
    const bf16* b0p = &Bs[mlane][k0 + 8 * hi];
    const bf16* b1p = &Bs[16 + mlane][k0 + 8 * hi];
    v16bf b0 = cat8(*(const v8bf*)b0p, *(const v8bf*)(b0p + 16));
    v16bf b1v = cat8(*(const v8bf*)b1p, *(const v8bf*)(b1p + 16));
    acc0 = __builtin_amdgcn_wmma_f32_16x16x32_bf16(
        false, a, false, b0, (short)0, acc0, false, false);
    acc1 = __builtin_amdgcn_wmma_f32_16x16x32_bf16(
        false, a, false, b1v, (short)0, acc1, false, false);
  }

  #pragma unroll
  for (int jt = 0; jt < 2; ++jt) {
    int ncol = jt * 16 + mlane;
    if (ncol >= NCLS) continue;
    float bias = b2[ncol];
    v8f acc = jt ? acc1 : acc0;
    #pragma unroll
    for (int r = 0; r < 8; ++r) {
      int gr = rowBase + r + 8 * hi;
      if (gr < N_NODES) h0[(size_t)gr * NCLS + ncol] = acc[r] + bias;
    }
  }
}

// ---------------------------------------------------------------------------
// Graph preprocessing + APPNP propagation (h is L2-resident: 7.6MB << 192MB)
// ---------------------------------------------------------------------------
__global__ void cvt_edges_kernel(const long long* __restrict__ ei,
                                 int* __restrict__ src, int* __restrict__ dst) {
  int e = blockIdx.x * blockDim.x + threadIdx.x;
  if (e >= N_EDGES) return;
  src[e] = (int)ei[e];
  dst[e] = (int)ei[(size_t)N_EDGES + e];
}

__global__ void fill_deg_kernel(int* __restrict__ degi) {
  int i = blockIdx.x * blockDim.x + threadIdx.x;
  if (i < N_NODES) degi[i] = 1;   // self-loop
}

__global__ void deg_count_kernel(const int* __restrict__ dst, int* __restrict__ degi) {
  int e = blockIdx.x * blockDim.x + threadIdx.x;
  if (e < N_EDGES) atomicAdd(&degi[dst[e]], 1);
}

__global__ void dinv_kernel(const int* __restrict__ degi, float* __restrict__ dinv) {
  int i = blockIdx.x * blockDim.x + threadIdx.x;
  if (i < N_NODES) dinv[i] = rsqrtf((float)degi[i]);
}

__global__ void norm_kernel(const int* __restrict__ src, const int* __restrict__ dst,
                            const float* __restrict__ dinv, float* __restrict__ norm) {
  int e = blockIdx.x * blockDim.x + threadIdx.x;
  if (e < N_EDGES) norm[e] = dinv[src[e]] * dinv[dst[e]];
}

// hnext = ALPHA*h0 + (1-ALPHA)*dinv[i]^2*hcur   (self-loop folded in)
__global__ void prop_init_kernel(const float* __restrict__ h0,
                                 const float* __restrict__ hcur,
                                 const float* __restrict__ dinv,
                                 float* __restrict__ hnext) {
  int idx = blockIdx.x * blockDim.x + threadIdx.x;
  if (idx >= N_NODES * NCLS) return;
  int i = idx / NCLS;
  float di = dinv[i];
  hnext[idx] = ALPHA_F * h0[idx] + (1.0f - ALPHA_F) * di * di * hcur[idx];
}

__global__ void scatter_kernel(const int* __restrict__ src, const int* __restrict__ dst,
                               const float* __restrict__ norm,
                               const float* __restrict__ hcur, float* __restrict__ hnext) {
  int e = blockIdx.x * blockDim.x + threadIdx.x;
  if (e >= N_EDGES) return;
  int s = src[e], d = dst[e];
  float wgt = (1.0f - ALPHA_F) * norm[e];
  const float* hs = hcur + (size_t)s * NCLS;
  float* hd = hnext + (size_t)d * NCLS;
  #pragma unroll
  for (int c = 0; c < NCLS; ++c) unsafeAtomicAdd(&hd[c], wgt * hs[c]);
}

// ---------------------------------------------------------------------------
extern "C" void kernel_launch(void* const* d_in, const int* in_sizes, int n_in,
                              void* d_out, int out_size, void* d_ws, size_t ws_size,
                              hipStream_t stream) {
  const float*     x  = (const float*)d_in[0];
  const long long* ei = (const long long*)d_in[1];
  const float*     W1 = (const float*)d_in[2];
  const float*     b1 = (const float*)d_in[3];
  const float*     W2 = (const float*)d_in[4];
  const float*     b2 = (const float*)d_in[5];
  float* out = (float*)d_out;

  char* ws = (char*)d_ws;
  size_t off = 0;
  auto carve = [&](size_t bytes) {
    void* p = ws + off;
    off += (bytes + 255) & ~(size_t)255;
    return p;
  };
  bf16*  h1   = (bf16*) carve((size_t)N_NODES * HIDF * sizeof(bf16));
  bf16*  w1t  = (bf16*) carve((size_t)F_INF * HIDF * sizeof(bf16));
  int*   src  = (int*)  carve((size_t)N_EDGES * sizeof(int));
  int*   dst  = (int*)  carve((size_t)N_EDGES * sizeof(int));
  float* norm = (float*)carve((size_t)N_EDGES * sizeof(float));
  int*   degi = (int*)  carve((size_t)N_NODES * sizeof(int));
  float* dinv = (float*)carve((size_t)N_NODES * sizeof(float));
  float* h0   = (float*)carve((size_t)N_NODES * NCLS * sizeof(float));
  float* bufA = (float*)carve((size_t)N_NODES * NCLS * sizeof(float));
  float* bufB = (float*)carve((size_t)N_NODES * NCLS * sizeof(float));

  // MLP (bf16 WMMA)
  cvt_w1_kernel<<<dim3((F_INF * HIDF + 255) / 256), dim3(256), 0, stream>>>(W1, w1t);
  gemm1_relu_kernel<<<dim3((N_NODES + 63) / 64), dim3(256), 0, stream>>>(x, w1t, b1, h1);
  gemm2_kernel<<<dim3((N_NODES + 127) / 128), dim3(256), 0, stream>>>(h1, W2, b2, h0);

  // Graph preprocessing (once)
  const int EB = (N_EDGES + 255) / 256, NB = (N_NODES + 255) / 256;
  cvt_edges_kernel<<<dim3(EB), dim3(256), 0, stream>>>(ei, src, dst);
  fill_deg_kernel<<<dim3(NB), dim3(256), 0, stream>>>(degi);
  deg_count_kernel<<<dim3(EB), dim3(256), 0, stream>>>(dst, degi);
  dinv_kernel<<<dim3(NB), dim3(256), 0, stream>>>(degi, dinv);
  norm_kernel<<<dim3(EB), dim3(256), 0, stream>>>(src, dst, dinv, norm);

  // K=10 APPNP iterations; last one writes d_out directly
  const int HB = (N_NODES * NCLS + 255) / 256;
  const float* cur = h0;
  for (int it = 0; it < KPROP; ++it) {
    float* nxt = (it == KPROP - 1) ? out : ((it % 2 == 0) ? bufA : bufB);
    prop_init_kernel<<<dim3(HB), dim3(256), 0, stream>>>(h0, cur, dinv, nxt);
    scatter_kernel<<<dim3(EB), dim3(256), 0, stream>>>(src, dst, norm, cur, nxt);
    cur = nxt;
  }
  (void)in_sizes; (void)n_in; (void)out_size; (void)ws_size;
}
